// Prompt_24678882082863
// MI455X (gfx1250) — matrix operand order repaired and checked by reading, not compiled, not run on hardware
//
#include <hip/hip_runtime.h>

typedef __attribute__((ext_vector_type(16))) __bf16 v16bf;
typedef __attribute__((ext_vector_type(8)))  float  v8f;

#define B_DIM 32
#define S_DIM 4096
#define D_DIM 768
#define V_DIM 500
#define V_PAD 512
#define ROWS (B_DIM * S_DIM)          // 131072
#define ROWS_PER_WG 64
#define NUM_WG (ROWS / ROWS_PER_WG)   // 2048
#define THREADS 128
#define LDS_STRIDE 776                // 768 + 8 pad -> conflict-free ds_load_b128
#define K_STEPS (D_DIM / 32)          // 24
#define EPSF 1e-12f

__device__ __forceinline__ unsigned short f2bf(float f) {
    unsigned u = __float_as_uint(f);
    u += 0x7FFFu + ((u >> 16) & 1u);  // round-to-nearest-even
    return (unsigned short)(u >> 16);
}

union FragA { v16bf v; uint4 q[2]; };
union FragB { v16bf v; uint4 q[2]; };

// ---------------------------------------------------------------------------
// Kernel 1: prompt_norm = l2norm(wte) in fp32 + bf16, padded to V_PAD rows.
// ---------------------------------------------------------------------------
__global__ __launch_bounds__(THREADS) void prompt_prep(
    const float* __restrict__ wte, float* __restrict__ pnF,
    unsigned short* __restrict__ pnB) {
    const int v = blockIdx.x;
    const int tid = threadIdx.x;
    const int wave = tid >> 5, lane = tid & 31;
    __shared__ float ssw[4];
    __shared__ float rn_s;
    const float* wr = wte + (size_t)v * D_DIM;
    float s = 0.f;
    if (v < V_DIM) {
        #pragma unroll
        for (int j = 0; j < 6; ++j) { float x = wr[tid + 128 * j]; s += x * x; }
    }
    #pragma unroll
    for (int off = 16; off >= 1; off >>= 1) s += __shfl_xor(s, off, 32);
    if (lane == 0) ssw[wave] = s;
    __syncthreads();
    if (tid == 0) rn_s = rsqrtf(fmaxf(ssw[0] + ssw[1] + ssw[2] + ssw[3], EPSF));
    __syncthreads();
    const float rn = rn_s;
    #pragma unroll
    for (int j = 0; j < 6; ++j) {
        int d = tid + 128 * j;
        float nv = (v < V_DIM) ? wr[d] * rn : 0.f;
        pnF[(size_t)v * D_DIM + d] = nv;
        pnB[(size_t)v * D_DIM + d] = f2bf(nv);
    }
}

// ---------------------------------------------------------------------------
// Kernel 2: fused normalize + WMMA similarity GEMM + argmax + gathers.
// 128 threads = 4 waves; 64 rows per workgroup; wave w owns rows [16w,16w+16).
// All large output streams use non-temporal stores so the 1 GB write stream
// does not evict the L2-resident bf16 prompt table / x tiles.
// ---------------------------------------------------------------------------
__global__ __launch_bounds__(THREADS) void prompt_main(
    const float* __restrict__ x, const float* __restrict__ wte,
    const float* __restrict__ pnF, const unsigned short* __restrict__ pnB,
    float* __restrict__ pe, float* __restrict__ sim,
    float* __restrict__ idxOut, float* __restrict__ bkn,
    float* __restrict__ wgsum) {
    __shared__ unsigned short Atile[ROWS_PER_WG * LDS_STRIDE];  // ~97 KB bf16
    __shared__ float ssw[4][ROWS_PER_WG];
    __shared__ float rmaxLds[ROWS_PER_WG];
    __shared__ int   ridxLds[ROWS_PER_WG];

    const int tid  = threadIdx.x;
    const int wave = tid >> 5;
    const int lane = tid & 31;
    const int hl   = lane >> 4;   // wave half (ISA bf16 fragment layout)
    const int l16  = lane & 15;
    const size_t rowBase = (size_t)blockIdx.x * ROWS_PER_WG;
    const float* xw = x + rowBase * D_DIM;

    // Pass 1: per-row sum-of-squares (coalesced; wave shfl-reduce)
    for (int r = 0; r < ROWS_PER_WG; ++r) {
        const float* xr = xw + (size_t)r * D_DIM;
        float s = 0.f;
        #pragma unroll
        for (int j = 0; j < 6; ++j) { float v = xr[tid + 128 * j]; s += v * v; }
        #pragma unroll
        for (int off = 16; off >= 1; off >>= 1) s += __shfl_xor(s, off, 32);
        if (lane == 0) ssw[wave][r] = s;
    }
    __syncthreads();

    // Pass 2: normalized bf16 A-tile in LDS (global reads hit WGP$/L2)
    for (int r = 0; r < ROWS_PER_WG; ++r) {
        float rn = rsqrtf(fmaxf(ssw[0][r] + ssw[1][r] + ssw[2][r] + ssw[3][r], EPSF));
        const float* xr = xw + (size_t)r * D_DIM;
        #pragma unroll
        for (int j = 0; j < 6; ++j) {
            int d = tid + 128 * j;
            Atile[r * LDS_STRIDE + d] = f2bf(xr[d] * rn);
        }
    }
    __syncthreads();

    const unsigned short* Abase = Atile + (wave * 16 + l16) * LDS_STRIDE;
    float runmax[8];
    int   runidx[8];
    #pragma unroll
    for (int r = 0; r < 8; ++r) { runmax[r] = -3.4e38f; runidx[r] = 0; }

    const v8f zero = {0.f, 0.f, 0.f, 0.f, 0.f, 0.f, 0.f, 0.f};

    for (int c = 0; c < V_PAD / 64; ++c) {         // 8 chunks of 64 columns
        v8f acc[4];
        #pragma unroll
        for (int t = 0; t < 4; ++t) acc[t] = zero;
        for (int k = 0; k < K_STEPS; ++k) {
            FragA a;  // A lane: K chunks [8h..8h+7], [16+8h..16+8h+7]
            const unsigned short* ap = Abase + k * 32 + 8 * hl;
            a.q[0] = *(const uint4*)(ap);
            a.q[1] = *(const uint4*)(ap + 16);
            #pragma unroll
            for (int t = 0; t < 4; ++t) {
                FragB b;  // B lane: K chunks [8h..8h+7], [16+8h..16+8h+7]
                const unsigned short* bp =
                    pnB + (size_t)(c * 64 + t * 16 + l16) * D_DIM + k * 32 + 8 * hl;
                b.q[0] = *(const uint4*)(bp);
                b.q[1] = *(const uint4*)(bp + 16);
                acc[t] = __builtin_amdgcn_wmma_f32_16x16x32_bf16(
                    false, a.v, false, b.v, (short)0, acc[t], false, false);
            }
        }
        // per-chunk epilogue: emit similarity (streaming), track running argmax
        #pragma unroll
        for (int t = 0; t < 4; ++t) {
            int n = c * 64 + t * 16 + l16;
            bool valid = (n < V_DIM);
            #pragma unroll
            for (int r = 0; r < 8; ++r) {
                float vv = acc[t][r];
                size_t grow = rowBase + wave * 16 + 8 * hl + r;  // C layout: M=r+8h
                if (valid) {
                    __builtin_nontemporal_store(vv, sim + grow * V_DIM + n);
                    if (vv > runmax[r]) { runmax[r] = vv; runidx[r] = n; }
                }
            }
        }
    }

    // cross-lane argmax within each 16-lane group (same rows, disjoint N)
    #pragma unroll
    for (int r = 0; r < 8; ++r) {
        float m = runmax[r];
        int   i = runidx[r];
        #pragma unroll
        for (int off = 8; off >= 1; off >>= 1) {
            float om = __shfl_xor(m, off, 32);
            int   oi = __shfl_xor(i, off, 32);
            if (om > m || (om == m && oi < i)) { m = om; i = oi; }
        }
        if (l16 == 0) {
            rmaxLds[wave * 16 + 8 * hl + r] = m;
            ridxLds[wave * 16 + 8 * hl + r] = i;
        }
    }
    __syncthreads();

    // deterministic per-WG partial for reduce_sim (= sum of per-row maxima)
    if (tid == 0) {
        float s = 0.f;
        for (int r = 0; r < ROWS_PER_WG; ++r) s += rmaxLds[r];
        wgsum[blockIdx.x] = s;
    }

    // gather epilogue: prompted_embedding, idx, batched_key_norm (streaming out)
    for (int r = 0; r < ROWS_PER_WG; ++r) {
        int v = ridxLds[r];
        size_t grow = rowBase + r;
        if (tid == 0) idxOut[grow] = (float)v;
        const float* wr = wte + (size_t)v * D_DIM;
        const float* pr = pnF + (size_t)v * D_DIM;
        const float* xr = xw + (size_t)r * D_DIM;
        #pragma unroll
        for (int j = 0; j < 6; ++j) {
            int d = tid + 128 * j;
            __builtin_nontemporal_store(wr[d] + xr[d], pe + grow * D_DIM + d);
            __builtin_nontemporal_store(pr[d], bkn + grow * D_DIM + d);
        }
    }
}

// ---------------------------------------------------------------------------
// Kernel 3: deterministic fixed-order reduction of per-WG partials.
// ---------------------------------------------------------------------------
__global__ __launch_bounds__(256) void reduce_sim_kernel(
    const float* __restrict__ wgsum, float* __restrict__ out) {
    __shared__ float sm[256];
    int tid = threadIdx.x;
    float s = 0.f;
    for (int i = tid; i < NUM_WG; i += 256) s += wgsum[i];
    sm[tid] = s;
    __syncthreads();
    for (int off = 128; off >= 1; off >>= 1) {
        if (tid < off) sm[tid] += sm[tid + off];
        __syncthreads();
    }
    if (tid == 0) *out = sm[0] / (float)B_DIM;  // / (B * TOP_K)
}

extern "C" void kernel_launch(void* const* d_in, const int* in_sizes, int n_in,
                              void* d_out, int out_size, void* d_ws, size_t ws_size,
                              hipStream_t stream) {
    (void)in_sizes; (void)n_in; (void)out_size; (void)ws_size;
    const float* x   = (const float*)d_in[0];   // [32,4096,768]
    const float* wte = (const float*)d_in[1];   // [500,768]
    float* out = (float*)d_out;

    // tuple layout: pe | similarity | reduce_sim | idx | batched_key_norm
    const size_t peOff  = 0;
    const size_t simOff = (size_t)ROWS * D_DIM;
    const size_t rsOff  = simOff + (size_t)ROWS * V_DIM;
    const size_t idxOff = rsOff + 1;
    const size_t bknOff = idxOff + (size_t)ROWS;

    float*          pnF   = (float*)d_ws;                              // 512*768 f32
    unsigned short* pnB   = (unsigned short*)(pnF + (size_t)V_PAD * D_DIM); // 512*768 bf16
    float*          wgsum = (float*)(pnB + (size_t)V_PAD * D_DIM);     // 2048 f32

    prompt_prep<<<V_PAD, THREADS, 0, stream>>>(wte, pnF, pnB);
    prompt_main<<<NUM_WG, THREADS, 0, stream>>>(
        x, wte, pnF, pnB, out + peOff, out + simOff, out + idxOff,
        out + bknOff, wgsum);
    reduce_sim_kernel<<<1, 256, 0, stream>>>(wgsum, out + rsOff);
}
